// GNN_63410897158370
// MI455X (gfx1250) — matrix operand order repaired and checked by reading, not compiled, not run on hardware
//
#include <hip/hip_runtime.h>

// ---------------------------------------------------------------------------
// GNN forward for MI455X (gfx1250, wave32).
//  - All dense matmuls: v_wmma_f32_16x16x32_bf16, one wave per 16x16 tile,
//    bf16 operands (converted once), fp32 accumulation.
//  - All segment ops: fp32 global atomics; accumulators fit in the 192MB L2.
// ---------------------------------------------------------------------------

typedef __attribute__((ext_vector_type(16))) __bf16         v16bf;
typedef __attribute__((ext_vector_type(8)))  float          v8f;
typedef __attribute__((ext_vector_type(8)))  unsigned short ushort8;

__device__ __forceinline__ unsigned short f2bf(float f) {
    unsigned u = __float_as_uint(f);
    u += 0x7fffu + ((u >> 16) & 1u);        // round-to-nearest-even
    return (unsigned short)(u >> 16);
}

union Frag { v16bf bf; ushort8 us[2]; };

// A-matrix 16x32 bf16 fragment: elems 0..7 = K[8*hi ..], elems 8..15 = K[16+8*hi ..]
__device__ __forceinline__ v16bf load_a(const unsigned short* __restrict__ row, int k, int hi) {
    Frag f;
    f.us[0] = *reinterpret_cast<const ushort8*>(row + k + 8 * hi);
    f.us[1] = *reinterpret_cast<const ushort8*>(row + k + 16 + 8 * hi);
    return f.bf;
}
// B-matrix 32x16 bf16 fragment from W^T[n][k]: elems 0..7 = K[8*hi ..], elems 8..15 = K[16+8*hi ..]
__device__ __forceinline__ v16bf load_b(const unsigned short* __restrict__ row, int k, int hi) {
    Frag f;
    f.us[0] = *reinterpret_cast<const ushort8*>(row + k + 8 * hi);
    f.us[1] = *reinterpret_cast<const ushort8*>(row + k + 16 + 8 * hi);
    return f.bf;
}

// D = A*BT^T [+ A2*BT2^T] + bias, optional ReLU; writes f32 and optional bf16 copy.
// One wave (32 threads) per 16x16 output tile. EXEC is all-ones (no divergence).
__global__ void __launch_bounds__(32)
gemm_bf16_wmma(const unsigned short* __restrict__ A,  const unsigned short* __restrict__ BT,
               const unsigned short* __restrict__ A2, const unsigned short* __restrict__ BT2,
               const float* __restrict__ bias, float* __restrict__ C,
               unsigned short* __restrict__ Cbf, int K, int Ncols, int relu)
{
    const int tm = blockIdx.x, tn = blockIdx.y;
    const int lane = threadIdx.x;
    const int r16 = lane & 15, hi = lane >> 4;
    const unsigned short* arow = A  + (size_t)(tm * 16 + r16) * K;
    const unsigned short* brow = BT + (size_t)(tn * 16 + r16) * K;
    v8f c = {};
    for (int k = 0; k < K; k += 32) {
        v16bf a = load_a(arow, k, hi);
        v16bf b = load_b(brow, k, hi);
        c = __builtin_amdgcn_wmma_f32_16x16x32_bf16(false, a, false, b, (short)0, c, false, false);
    }
    if (A2) {
        const unsigned short* arow2 = A2  + (size_t)(tm * 16 + r16) * K;
        const unsigned short* brow2 = BT2 + (size_t)(tn * 16 + r16) * K;
        for (int k = 0; k < K; k += 32) {
            v16bf a = load_a(arow2, k, hi);
            v16bf b = load_b(brow2, k, hi);
            c = __builtin_amdgcn_wmma_f32_16x16x32_bf16(false, a, false, b, (short)0, c, false, false);
        }
    }
    const int col = tn * 16 + r16;
    const float bv = bias ? bias[col] : 0.0f;
#pragma unroll
    for (int r = 0; r < 8; ++r) {
        const int m = tm * 16 + r + 8 * hi;
        float v = c[r] + bv;
        if (relu) v = fmaxf(v, 0.0f);
        C[(size_t)m * Ncols + col] = v;
        if (Cbf) Cbf[(size_t)m * Ncols + col] = f2bf(v);
    }
}

// ------------------------------ utility kernels -----------------------------

__global__ void fill_u32(unsigned* __restrict__ p, unsigned v, long long n) {
    long long t = (long long)blockIdx.x * blockDim.x + threadIdx.x;
    if (t < n) p[t] = v;
}

__global__ void f32_to_bf16(const float* __restrict__ in, unsigned short* __restrict__ out, long long n) {
    long long t = (long long)blockIdx.x * blockDim.x + threadIdx.x;
    if (t < n) out[t] = f2bf(in[t]);
}

// W[K][Ncols] f32 -> WT[n][k] bf16 (transposed for the WMMA B-operand path)
__global__ void wT_bf16(const float* __restrict__ W, unsigned short* __restrict__ WT, int K, int Ncols) {
    int t = blockIdx.x * blockDim.x + threadIdx.x;
    if (t >= K * Ncols) return;
    int k = t / Ncols, n = t % Ncols;
    WT[(size_t)n * K + k] = f2bf(W[t]);
}

__global__ void deg_count(const int* __restrict__ dst, float* __restrict__ deg, int E) {
    int t = blockIdx.x * blockDim.x + threadIdx.x;
    if (t < E) atomicAdd(&deg[dst[t]], 1.0f);
}

// agg[dst] += x[src], 4 features per thread (gather from HBM, atomics land in L2)
__global__ void sage_scatter(const int* __restrict__ src, const int* __restrict__ dst,
                             const float* __restrict__ x, float* __restrict__ agg,
                             int F, long long total)
{
    long long t = (long long)blockIdx.x * blockDim.x + threadIdx.x;
    if (t >= total) return;
    const int q4 = F >> 2;
    const int e = (int)(t / q4);
    const int f = (int)(t % q4) * 4;
    const int s = src[e], d = dst[e];
    const float4 v = *(const float4*)(x + (size_t)s * F + f);
    float* ap = agg + (size_t)d * F + f;
    atomicAdd(ap + 0, v.x); atomicAdd(ap + 1, v.y);
    atomicAdd(ap + 2, v.z); atomicAdd(ap + 3, v.w);
}

__global__ void agg_mean_bf(const float* __restrict__ agg, const float* __restrict__ deg,
                            unsigned short* __restrict__ out, long long total, int F)
{
    long long t = (long long)blockIdx.x * blockDim.x + threadIdx.x;
    if (t >= total) return;
    float dv = deg[t / F];
    dv = dv < 1.0f ? 1.0f : dv;
    out[t] = f2bf(agg[t] / dv);
}

// per-node per-head attention coefficients: als/ald[n,h] = <xh[n,h,:], a_{src/dst}[h,:]>
__global__ void gat_alpha_coef(const float* __restrict__ xh, const float* __restrict__ a_s,
                               const float* __restrict__ a_d, float* __restrict__ als,
                               float* __restrict__ ald, int n, int c)
{
    int t = blockIdx.x * blockDim.x + threadIdx.x;
    if (t >= n * 4) return;
    const int node = t >> 2, h = t & 3;
    const float* xp = xh + (size_t)node * 4 * c + h * c;
    const float* sp = a_s + h * c;
    const float* dp = a_d + h * c;
    float s = 0.f, d = 0.f;
    for (int j = 0; j < c; ++j) { float v = xp[j]; s += v * sp[j]; d += v * dp[j]; }
    als[t] = s; ald[t] = d;
}

// leaky-relu logits + segment-max via order-preserving uint key atomicMax
__global__ void gat_edge_logits(const int* __restrict__ src, const int* __restrict__ dst,
                                const float* __restrict__ als, const float* __restrict__ ald,
                                float* __restrict__ ebuf, unsigned* __restrict__ mkey, long long total)
{
    long long t = (long long)blockIdx.x * blockDim.x + threadIdx.x;
    if (t >= total) return;
    const int e = (int)(t >> 2), h = (int)(t & 3);
    const int s = src[e], d = dst[e];
    float v = als[s * 4 + h] + ald[d * 4 + h];
    v = v > 0.f ? v : 0.2f * v;
    ebuf[t] = v;
    unsigned k = __float_as_uint(v);
    k = (k & 0x80000000u) ? ~k : (k | 0x80000000u);
    atomicMax(&mkey[d * 4 + h], k);
}

// exp(e - max) + segment-sum
__global__ void gat_edge_soft(const int* __restrict__ dst, const unsigned* __restrict__ mkey,
                              float* __restrict__ ebuf, float* __restrict__ ssum, long long total)
{
    long long t = (long long)blockIdx.x * blockDim.x + threadIdx.x;
    if (t >= total) return;
    const int e = (int)(t >> 2), h = (int)(t & 3);
    const int d = dst[e];
    unsigned k = mkey[d * 4 + h];
    float m = __uint_as_float((k & 0x80000000u) ? (k & 0x7fffffffu) : ~k);
    float ex = __expf(ebuf[t] - m);
    ebuf[t] = ex;
    atomicAdd(&ssum[d * 4 + h], ex);
}

// acc[dst] += xh[src] * alpha, 4 features per thread
__global__ void gat_scatter(const int* __restrict__ src, const int* __restrict__ dst,
                            const float* __restrict__ ebuf, const float* __restrict__ ssum,
                            const float* __restrict__ xh, float* __restrict__ acc,
                            int HC, int c, long long total)
{
    long long t = (long long)blockIdx.x * blockDim.x + threadIdx.x;
    if (t >= total) return;
    const int q4 = HC >> 2;
    const int e = (int)(t / q4);
    const int f = (int)(t % q4) * 4;
    const int h = f / c;
    const int s = src[e], d = dst[e];
    const float alpha = ebuf[(size_t)e * 4 + h] / (ssum[d * 4 + h] + 1e-16f);
    const float4 v = *(const float4*)(xh + (size_t)s * HC + f);
    float* ap = acc + (size_t)d * HC + f;
    atomicAdd(ap + 0, v.x * alpha); atomicAdd(ap + 1, v.y * alpha);
    atomicAdd(ap + 2, v.z * alpha); atomicAdd(ap + 3, v.w * alpha);
}

// bias + relu + head-mean; writes both f32 (for next scatter) and bf16 (next WMMA)
__global__ void gat_finish(const float* __restrict__ acc, const float* __restrict__ ba,
                           float* __restrict__ xout, unsigned short* __restrict__ xbf, int n, int c)
{
    int t = blockIdx.x * blockDim.x + threadIdx.x;
    if (t >= n * c) return;
    const int node = t / c, j = t % c;
    const float* ap = acc + (size_t)node * 4 * c;
    float s = 0.f;
#pragma unroll
    for (int h = 0; h < 4; ++h) s += fmaxf(ap[h * c + j] + ba[h * c + j], 0.0f);
    s *= 0.25f;
    xout[t] = s;
    xbf[t] = f2bf(s);
}

// penalty head + exp scale + L2 row-normalize: OUT=32 features == 32 lanes of one wave
__global__ void __launch_bounds__(256)
final_head(const float* __restrict__ x, const float* __restrict__ Wp,
           const float* __restrict__ bp, float* __restrict__ out, int n)
{
    const int w = blockIdx.x * (blockDim.x >> 5) + (threadIdx.x >> 5);
    const int lane = threadIdx.x & 31;
    if (w >= n) return;
    const float xv = x[(size_t)w * 32 + lane];
    float p = xv * Wp[lane];
#pragma unroll
    for (int o = 16; o; o >>= 1) p += __shfl_xor(p, o, 32);
    p += bp[0];
    const float y = xv * __expf(p);
    float s = y * y;
#pragma unroll
    for (int o = 16; o; o >>= 1) s += __shfl_xor(s, o, 32);
    const float norm = fmaxf(sqrtf(s), 1e-12f);
    out[(size_t)w * 32 + lane] = y / norm;
}

// --------------------------------- host ------------------------------------

static inline int div_up(long long a, long long b) { return (int)((a + b - 1) / b); }

extern "C" void kernel_launch(void* const* d_in, const int* in_sizes, int n_in,
                              void* d_out, int out_size, void* d_ws, size_t ws_size,
                              hipStream_t stream)
{
    const float* x_in = (const float*)d_in[0];
    const int*   ei   = (const int*)d_in[1];
    const int N = in_sizes[0] / 128;          // 100000 (divisible by 16)
    const int E = in_sizes[1] / 2;            // 1600000
    const int* src = ei;
    const int* dst = ei + E;

    const float* Wl[3]  = {(const float*)d_in[2],  (const float*)d_in[9],  (const float*)d_in[16]};
    const float* bl[3]  = {(const float*)d_in[3],  (const float*)d_in[10], (const float*)d_in[17]};
    const float* Wr[3]  = {(const float*)d_in[4],  (const float*)d_in[11], (const float*)d_in[18]};
    const float* Wa[3]  = {(const float*)d_in[5],  (const float*)d_in[12], (const float*)d_in[19]};
    const float* asv[3] = {(const float*)d_in[6],  (const float*)d_in[13], (const float*)d_in[20]};
    const float* adv[3] = {(const float*)d_in[7],  (const float*)d_in[14], (const float*)d_in[21]};
    const float* ba[3]  = {(const float*)d_in[8],  (const float*)d_in[15], (const float*)d_in[22]};
    const float* Wp = (const float*)d_in[23];
    const float* bp = (const float*)d_in[24];

    // ---- workspace layout (256B aligned regions) ----
    char* base = (char*)d_ws;
    size_t off = 0;
    auto take = [&](size_t bytes) -> char* {
        char* p = base + off;
        off += (bytes + 255) & ~(size_t)255;
        return p;
    };
    float*          deg  = (float*)take((size_t)N * 4);
    unsigned short* xbf  = (unsigned short*)take((size_t)N * 128 * 2);
    float*          x32  = (float*)take((size_t)N * 64 * 4);
    // big aliased region: [agg(N*128 f32) | aggbf(N*128 bf16) | h32(N*64 f32)] == acc(N*256 f32)
    // (agg/aggbf/h32 are all dead before acc is zero-filled; next block re-fills agg after gat_finish)
    char*           big  = take((size_t)N * 256 * 4);
    float*          agg  = (float*)big;
    unsigned short* aggbf= (unsigned short*)(big + (size_t)N * 128 * 4);
    float*          h32  = (float*)(big + (size_t)N * 128 * 4 + (size_t)N * 128 * 2);
    float*          acc  = (float*)big;
    unsigned short* hbf  = (unsigned short*)take((size_t)N * 64 * 2);
    float*          xh   = (float*)take((size_t)N * 256 * 4);
    float*          als  = (float*)take((size_t)N * 4 * 4);
    float*          ald  = (float*)take((size_t)N * 4 * 4);
    unsigned*       mkey = (unsigned*)take((size_t)N * 4 * 4);
    float*          ssum = (float*)take((size_t)N * 4 * 4);
    float*          ebuf = (float*)take((size_t)E * 4 * 4);
    unsigned short* wt   = (unsigned short*)take((size_t)65536 * 2);

    // transposed bf16 weights
    unsigned short *WlT[3], *WrT[3], *WaT[3];
    size_t wo = 0;
    const int WK[3]  = {128, 64, 64};     // SAGE input dims
    const int WS[3]  = {64, 64, 32};      // SAGE output dims
    const int WAK[3] = {64, 64, 32};      // GAT input dims
    const int WAN[3] = {256, 256, 128};   // GAT output dims (H*c)
    for (int b = 0; b < 3; ++b) {
        WlT[b] = wt + wo; wo += (size_t)WK[b] * WS[b];
        WrT[b] = wt + wo; wo += (size_t)WK[b] * WS[b];
        WaT[b] = wt + wo; wo += (size_t)WAK[b] * WAN[b];
    }

    const int TB = 256;

    // degrees (shared by all 3 SAGE layers)
    fill_u32<<<div_up(N, TB), TB, 0, stream>>>((unsigned*)deg, 0u, N);
    deg_count<<<div_up(E, TB), TB, 0, stream>>>(dst, deg, E);

    // weight prep
    for (int b = 0; b < 3; ++b) {
        wT_bf16<<<div_up((long long)WK[b] * WS[b], TB), TB, 0, stream>>>(Wl[b], WlT[b], WK[b], WS[b]);
        wT_bf16<<<div_up((long long)WK[b] * WS[b], TB), TB, 0, stream>>>(Wr[b], WrT[b], WK[b], WS[b]);
        wT_bf16<<<div_up((long long)WAK[b] * WAN[b], TB), TB, 0, stream>>>(Wa[b], WaT[b], WAK[b], WAN[b]);
    }

    // input features -> bf16
    f32_to_bf16<<<div_up((long long)N * 128, TB), TB, 0, stream>>>(x_in, xbf, (long long)N * 128);

    const float* curx = x_in;
    int Fin = 128;
    for (int b = 0; b < 3; ++b) {
        const int Fs = WS[b];          // SAGE out dim
        const int c  = WAK[b];         // per-head dim (== Fs)
        const int HC = WAN[b];         // H * c

        // ---- SAGE: mean-aggregate + two WMMA GEMMs + bias + relu ----
        const long long nf = (long long)N * Fin;
        fill_u32<<<div_up(nf, TB), TB, 0, stream>>>((unsigned*)agg, 0u, nf);
        const long long st = (long long)E * (Fin / 4);
        sage_scatter<<<div_up(st, TB), TB, 0, stream>>>(src, dst, curx, agg, Fin, st);
        agg_mean_bf<<<div_up(nf, TB), TB, 0, stream>>>(agg, deg, aggbf, nf, Fin);
        {
            dim3 g(N / 16, Fs / 16);
            gemm_bf16_wmma<<<g, 32, 0, stream>>>(aggbf, WlT[b], xbf, WrT[b], bl[b],
                                                 h32, hbf, Fin, Fs, 1);
        }

        // ---- GAT: xh = h @ Wa (WMMA), attention softmax, weighted scatter ----
        {
            dim3 g(N / 16, HC / 16);
            gemm_bf16_wmma<<<g, 32, 0, stream>>>(hbf, WaT[b], nullptr, nullptr, nullptr,
                                                 xh, nullptr, Fs, HC, 0);
        }
        gat_alpha_coef<<<div_up((long long)N * 4, TB), TB, 0, stream>>>(xh, asv[b], adv[b], als, ald, N, c);
        fill_u32<<<div_up((long long)N * 4, TB), TB, 0, stream>>>(mkey, 0u, (long long)N * 4);
        fill_u32<<<div_up((long long)N * 4, TB), TB, 0, stream>>>((unsigned*)ssum, 0u, (long long)N * 4);
        const long long et = (long long)E * 4;
        gat_edge_logits<<<div_up(et, TB), TB, 0, stream>>>(src, dst, als, ald, ebuf, mkey, et);
        gat_edge_soft<<<div_up(et, TB), TB, 0, stream>>>(dst, mkey, ebuf, ssum, et);
        const long long at = (long long)N * HC;
        fill_u32<<<div_up(at, TB), TB, 0, stream>>>((unsigned*)acc, 0u, at);
        const long long st3 = (long long)E * (HC / 4);
        gat_scatter<<<div_up(st3, TB), TB, 0, stream>>>(src, dst, ebuf, ssum, xh, acc, HC, c, st3);
        gat_finish<<<div_up((long long)N * c, TB), TB, 0, stream>>>(acc, ba[b], x32, xbf, N, c);

        curx = x32;
        Fin = c;
    }

    // ---- penalty head + exp scaling + row L2 normalize ----
    final_head<<<div_up(N, 8), 256, 0, stream>>>(x32, Wp, bp, (float*)d_out, N);
}
